// MultiHeadSelfAttention_56891136803414
// MI455X (gfx1250) — matrix operand order repaired and checked by reading, not compiled, not run on hardware
//
#include <hip/hip_runtime.h>

typedef __attribute__((ext_vector_type(16))) _Float16 v16h;
typedef __attribute__((ext_vector_type(8)))  _Float16 v8h;
typedef __attribute__((ext_vector_type(8)))  float    v8f;

#define B_SZ   2048
#define NTOK   50
#define CDIM   256
#define NHEAD  8
#define HDIM   32
#define MROWS  (B_SZ * NTOK)          /* 102400, divisible by 32 */
#define QSCALE 0.17677669529663689f   /* 1/sqrt(32) */

static __device__ __forceinline__ int lane_id() { return (int)(threadIdx.x & 31); }

// ---- CDNA5 async global->LDS copy (ASYNCcnt path, ISA 08 section 4) ----
// LDS address operand = low 32 bits of the generic pointer (LDS aperture
// truncates to addr[31:0], relative to the wave's LDS base).
static __device__ __forceinline__ void async_copy_b128(void* lds_dst, const void* gsrc) {
  const unsigned l = (unsigned)(unsigned long long)(uintptr_t)lds_dst;
  const unsigned long long g = (unsigned long long)(uintptr_t)gsrc;
  asm volatile("global_load_async_to_lds_b128 %0, %1, off"
               :: "v"(l), "v"(g) : "memory");
}
static __device__ __forceinline__ void wait_async0() {
  asm volatile("s_wait_asynccnt 0x0" ::: "memory");
}

// ---- WMMA fragment loaders (contiguous k order k = 16*(lane/16) + s in A and B, wave32) ----

// A fragment 16x32 f16, row-major source, row stride in halves.
// lane<16: row=lane, K={0..15}; lane>=16: row=lane-16, K={16..31}
static __device__ __forceinline__ v16h frag_a_f16(const _Float16* src, int stride) {
  const int lane = lane_id();
  const int row  = lane & 15;
  const int kb   = (lane >> 4) << 4;              // 0 or 16
  const _Float16* p = src + row * stride + kb;
  union { v16h v; v8h h[2]; } u;
  u.h[0] = *(const v8h*)(p);                      // K kb..kb+7
  u.h[1] = *(const v8h*)(p + 8);                  // K kb+8..kb+15
  return u.v;
}

// B fragment 32x16 f16 where source is column-major-per-token: element(k,n)=src[n*stride+k]
static __device__ __forceinline__ v16h frag_b_f16_kcontig(const _Float16* src, int stride) {
  const int lane = lane_id();
  const int col  = lane & 15;
  const int kb   = (lane >> 4) << 4;              // 0 or 16
  const _Float16* p = src + col * stride + kb;
  union { v16h v; v8h h[2]; } u;
  u.h[0] = *(const v8h*)(p);
  u.h[1] = *(const v8h*)(p + 8);
  return u.v;
}

// B fragment 32x16 f16 where source is row-major over K: element(k,n)=src[k*stride+n]
static __device__ __forceinline__ v16h frag_b_f16_kstrided(const _Float16* src, int stride) {
  const int lane = lane_id();
  const int col  = lane & 15;
  const int kb   = (lane >> 4) << 4;
  v16h b;
#pragma unroll
  for (int i = 0; i < 16; ++i) b[i] = src[(kb + i) * stride + col];
  return b;
}

// B fragment from pre-converted f16 weight matrix W[n][k] (row-major, ld=K)
static __device__ __forceinline__ v16h frag_b_w_f16(const _Float16* w, int ld, int n0, int k0) {
  const int lane = lane_id();
  const int col  = lane & 15;
  const int kb   = ((lane >> 4) << 4) + k0;
  const _Float16* p = w + (size_t)(n0 + col) * ld + kb;
  union { v16h v; v8h h[2]; } u;
  u.h[0] = *(const v8h*)(p);
  u.h[1] = *(const v8h*)(p + 8);
  return u.v;
}

// ---- Kernel 0: f32 -> f16 weight conversion (run once per weight matrix) ----
__global__ void cvt_f32_f16_kernel(const float* __restrict__ src,
                                   _Float16* __restrict__ dst, int n) {
  const int i = blockIdx.x * blockDim.x + threadIdx.x;
  if (i < n) dst[i] = (_Float16)src[i];
}

// ---- Kernel 1: expand padded relative-position bias to [8][64][64] f32 ----
__global__ void bias_expand_kernel(const float* __restrict__ table,
                                   const int* __restrict__ rel_idx,
                                   float* __restrict__ biasfull) {
  const int h = blockIdx.x;
  for (int idx = threadIdx.x; idx < 64 * 64; idx += blockDim.x) {
    const int i = idx >> 6, j = idx & 63;
    float v = 0.f;
    if (i >= 1 && i < NTOK && j >= 1 && j < NTOK)
      v = table[rel_idx[(i - 1) * 49 + (j - 1)] * NHEAD + h];
    biasfull[h * 4096 + idx] = v;
  }
}

// ---- Kernel 2: QKV GEMM: [102400,256] x [256,768]^T -> q/k/v f16 [B,H,N,hd] ----
// Block tile 32(M) x 256(N): 8 waves in 2x4; each wave: 16x64 (4 accumulators).
__global__ void __launch_bounds__(256)
qkv_gemm_kernel(const float* __restrict__ x, const _Float16* __restrict__ w16,
                const float* __restrict__ qkv_b,
                _Float16* __restrict__ qf, _Float16* __restrict__ kf,
                _Float16* __restrict__ vf) {
  __shared__ __align__(16) _Float16 aS[32 * 256];
  const int m0 = blockIdx.x * 32;
  const int n0 = blockIdx.y * 256;

  // Warm the weight slice for this block's columns (global_prefetch_b8 path).
  __builtin_prefetch(w16 + ((size_t)n0 + threadIdx.x) * CDIM, 0, 3);

  {
    const float* src = x + (size_t)m0 * CDIM;
#pragma unroll
    for (int i = 0; i < 32; ++i) {
      const int idx = (int)threadIdx.x * 32 + i;
      aS[idx] = (_Float16)src[idx];
    }
  }
  __syncthreads();

  const int w    = threadIdx.x >> 5;
  const int wm   = w >> 2;                 // 0..1 : M sub-tile
  const int wn   = w & 3;                  // 0..3 : N strip of 64
  const int lane = threadIdx.x & 31;
  const int col  = lane & 15;
  const int half = lane >> 4;
  const int nb   = n0 + wn * 64;

  v8f c[4] = {};
#pragma unroll
  for (int k0 = 0; k0 < 256; k0 += 32) {
    v16h a = frag_a_f16(aS + wm * (16 * 256) + k0, 256);
#pragma unroll
    for (int t = 0; t < 4; ++t) {
      v16h b = frag_b_w_f16(w16, 256, nb + t * 16, k0);
      c[t] = __builtin_amdgcn_wmma_f32_16x16x32_f16(false, a, false, b, (short)0,
                                                    c[t], false, false);
    }
  }

#pragma unroll
  for (int t = 0; t < 4; ++t) {
    const int   cg    = nb + t * 16 + col;
    const int   which = cg >> 8;            // 0=q 1=k 2=v (uniform per wave strip)
    const int   rem   = cg & 255;
    const int   head  = rem >> 5;
    const int   dim   = rem & 31;
    const float bias  = qkv_b[cg];
    const float mul   = (which == 0) ? QSCALE : 1.0f;
    _Float16* dst = (which == 0) ? qf : (which == 1) ? kf : vf;
#pragma unroll
    for (int r = 0; r < 8; ++r) {
      const int m  = m0 + wm * 16 + half * 8 + r;
      const int bb = m / NTOK, nn = m - bb * NTOK;
      dst[(((size_t)(bb * NHEAD + head) * NTOK + nn) * HDIM) + dim] =
          (_Float16)((c[t][r] + bias) * mul);
    }
  }
}

// ---- Kernel 3: attention per (b,h): scores + bias + softmax + P*V ----
__global__ void __launch_bounds__(128)
attn_kernel(const _Float16* __restrict__ qf, const _Float16* __restrict__ kf,
            const _Float16* __restrict__ vf, const float* __restrict__ biasfull,
            _Float16* __restrict__ ao) {
  __shared__ __align__(16) _Float16 qS[64 * 32];
  __shared__ __align__(16) _Float16 kS[64 * 32];
  __shared__ __align__(16) _Float16 vS[64 * 32];
  __shared__ float    sS[64 * 65];    // stride 65 -> bank-conflict-free row sweeps
  __shared__ _Float16 pS[64 * 64];

  const int bh  = blockIdx.x;         // b*8 + h
  const int h   = bh & 7;
  const int bb  = bh >> 3;
  const int tid = threadIdx.x;
  const size_t base = (size_t)bh * (NTOK * HDIM);

  // Zero-fill pad rows [50,64) with regular ds stores.
  for (int j = NTOK * HDIM + tid; j < 64 * HDIM; j += 128) {
    qS[j] = (_Float16)0.f; kS[j] = (_Float16)0.f; vS[j] = (_Float16)0.f;
  }
  // Async-copy the 3200 valid bytes of each of q/k/v straight into LDS
  // (ASYNCcnt-tracked, bypasses VGPRs).
  {
    const char* gq = (const char*)(qf + base);
    const char* gk = (const char*)(kf + base);
    const char* gv = (const char*)(vf + base);
    for (int off = tid * 16; off < NTOK * HDIM * 2; off += 128 * 16) {
      async_copy_b128((char*)qS + off, gq + off);
      async_copy_b128((char*)kS + off, gk + off);
      async_copy_b128((char*)vS + off, gv + off);
    }
    wait_async0();
  }
  __syncthreads();

  const int w    = tid >> 5;
  const int lane = tid & 31;
  const int col  = lane & 15;
  const int half = lane >> 4;

  // scores rows [16w,16w+16) x cols [0,64): S = q * k^T
  {
    v16h a = frag_a_f16(qS + (w * 16) * 32, 32);
#pragma unroll
    for (int nt = 0; nt < 4; ++nt) {
      v16h b = frag_b_f16_kcontig(kS + (nt * 16) * 32, 32);
      v8f cfr = {};
      cfr = __builtin_amdgcn_wmma_f32_16x16x32_f16(false, a, false, b, (short)0,
                                                   cfr, false, false);
#pragma unroll
      for (int r = 0; r < 8; ++r)
        sS[(w * 16 + half * 8 + r) * 65 + nt * 16 + col] = cfr[r];
    }
  }
  __syncthreads();

  // per-row softmax over 50 valid columns, +bias, write f16 probs (padded = 0)
  if (tid < 64) {
    const int i = tid;
    float*    srow = &sS[i * 65];
    _Float16* prow = &pS[i * 64];
    if (i < NTOK) {
      const float* brow = biasfull + (h * 64 + i) * 64;
      float mx = -1e30f;
      for (int j = 0; j < NTOK; ++j) {
        const float v = srow[j] + brow[j];
        srow[j] = v;
        mx = fmaxf(mx, v);
      }
      float sum = 0.f;
      for (int j = 0; j < NTOK; ++j) {
        const float e = __expf(srow[j] - mx);
        srow[j] = e;
        sum += e;
      }
      const float inv = 1.0f / sum;
      for (int j = 0; j < NTOK; ++j) prow[j] = (_Float16)(srow[j] * inv);
      for (int j = NTOK; j < 64; ++j) prow[j] = (_Float16)0.f;
    } else {
      for (int j = 0; j < 64; ++j) prow[j] = (_Float16)0.f;
    }
  }
  __syncthreads();

  // out rows [16w,16w+16) x dims [0,32): O = P * V  (K=64, two steps)
#pragma unroll
  for (int nt = 0; nt < 2; ++nt) {
    v8f cfr = {};
#pragma unroll
    for (int ks = 0; ks < 2; ++ks) {
      v16h a = frag_a_f16(pS + (w * 16) * 64 + ks * 32, 64);
      v16h b = frag_b_f16_kstrided(vS + (ks * 32) * 32 + nt * 16, 32);
      cfr = __builtin_amdgcn_wmma_f32_16x16x32_f16(false, a, false, b, (short)0,
                                                   cfr, false, false);
    }
#pragma unroll
    for (int r = 0; r < 8; ++r) {
      const int m = w * 16 + half * 8 + r;
      if (m < NTOK)
        ao[((size_t)bb * NTOK + m) * CDIM + h * HDIM + nt * 16 + col] = (_Float16)cfr[r];
    }
  }
}

// ---- Kernel 4: output projection: [102400,256] x [256,256]^T + bias -> f32 ----
// Block tile 32(M) x 256(N): 8 waves in 2x4; each wave: 16x64 (4 accumulators).
__global__ void __launch_bounds__(256)
proj_gemm_kernel(const _Float16* __restrict__ ao, const _Float16* __restrict__ w16,
                 const float* __restrict__ proj_b, float* __restrict__ out) {
  __shared__ __align__(16) _Float16 aS[32 * 256];
  const int m0 = blockIdx.x * 32;

  __builtin_prefetch(w16 + (size_t)threadIdx.x * CDIM, 0, 3);

  // Async-stage the 16KB A tile (f16) straight into LDS: 4 x 16B per thread.
  {
    const char* src = (const char*)(ao + (size_t)m0 * CDIM);
#pragma unroll
    for (int it = 0; it < 4; ++it) {
      const int off = ((int)threadIdx.x + it * 256) * 16;
      async_copy_b128((char*)aS + off, src + off);
    }
    wait_async0();
  }
  __syncthreads();

  const int w    = threadIdx.x >> 5;
  const int wm   = w >> 2;
  const int wn   = w & 3;
  const int lane = threadIdx.x & 31;
  const int col  = lane & 15;
  const int half = lane >> 4;
  const int nb   = wn * 64;

  v8f c[4] = {};
#pragma unroll
  for (int k0 = 0; k0 < 256; k0 += 32) {
    v16h a = frag_a_f16(aS + wm * (16 * 256) + k0, 256);
#pragma unroll
    for (int t = 0; t < 4; ++t) {
      v16h b = frag_b_w_f16(w16, 256, nb + t * 16, k0);
      c[t] = __builtin_amdgcn_wmma_f32_16x16x32_f16(false, a, false, b, (short)0,
                                                    c[t], false, false);
    }
  }

#pragma unroll
  for (int t = 0; t < 4; ++t) {
    const int   cg   = nb + t * 16 + col;
    const float bias = proj_b[cg];
#pragma unroll
    for (int r = 0; r < 8; ++r) {
      const int m = m0 + wm * 16 + half * 8 + r;
      out[(size_t)m * CDIM + cg] = c[t][r] + bias;
    }
  }
}

extern "C" void kernel_launch(void* const* d_in, const int* in_sizes, int n_in,
                              void* d_out, int out_size, void* d_ws, size_t ws_size,
                              hipStream_t stream) {
  (void)in_sizes; (void)n_in; (void)out_size; (void)ws_size;
  const float* x       = (const float*)d_in[0];
  const float* qkv_w   = (const float*)d_in[1];
  const float* qkv_b   = (const float*)d_in[2];
  const float* proj_w  = (const float*)d_in[3];
  const float* proj_b  = (const float*)d_in[4];
  const float* table   = (const float*)d_in[5];
  const int*   rel_idx = (const int*)d_in[6];
  float* out = (float*)d_out;

  const size_t NQ = (size_t)B_SZ * NHEAD * NTOK * HDIM;   // 26,214,400 halves each
  _Float16* qf = (_Float16*)d_ws;
  _Float16* kf = qf + NQ;
  _Float16* vf = kf + NQ;
  _Float16* ao = vf + NQ;                                  // [B, N, C] f16
  float* biasfull = (float*)(ao + (size_t)MROWS * CDIM);   // [8][64][64] f32
  _Float16* qkv_w16  = (_Float16*)(biasfull + NHEAD * 64 * 64);  // [768][256]
  _Float16* proj_w16 = qkv_w16 + (size_t)(3 * CDIM) * CDIM;      // [256][256]

  cvt_f32_f16_kernel<<<(3 * CDIM * CDIM) / 256, 256, 0, stream>>>(
      qkv_w, qkv_w16, 3 * CDIM * CDIM);
  cvt_f32_f16_kernel<<<(CDIM * CDIM) / 256, 256, 0, stream>>>(
      proj_w, proj_w16, CDIM * CDIM);
  bias_expand_kernel<<<NHEAD, 256, 0, stream>>>(table, rel_idx, biasfull);
  qkv_gemm_kernel<<<dim3(MROWS / 32, (3 * CDIM) / 256), 256, 0, stream>>>(
      x, qkv_w16, qkv_b, qf, kf, vf);
  attn_kernel<<<B_SZ * NHEAD, 128, 0, stream>>>(qf, kf, vf, biasfull, ao);
  proj_gemm_kernel<<<dim3(MROWS / 32, 1), 256, 0, stream>>>(
      ao, proj_w16, proj_b, out);
}
